// REFINE_74182675136889
// MI455X (gfx1250) — matrix operand order repaired and checked by reading, not compiled, not run on hardware
//
#include <hip/hip_runtime.h>
#include <hip/hip_bf16.h>

typedef __bf16 bf16;
typedef __attribute__((ext_vector_type(16))) __bf16 v16bf;
typedef __attribute__((ext_vector_type(8)))  __bf16 v8bf;
typedef __attribute__((ext_vector_type(8)))  float   v8f;

#define LDS_STRIDE 520          // 512 + 8 bf16 pad -> 1040B row, 4-bank skew per row
#define M_TILE     128          // rows per block = 8 WMMA row-tiles (2x133KB LDS total)
#define NRT        8            // row tiles per block
#define OUT_STRIDE 1114112      // floats per batch in d_out (16384*64 + 32768*2)
#define LAT_OFF    1048576      // occ region starts after 16384*64 latent floats
#define INV_2PI    0.15915494309189535f

// ---------------- fragment loaders ----------------
// A (16x32 bf16, M=lane%16): element i -> K = (i<8?0:16) + (lane/16)*8 + (i%8)
__device__ __forceinline__ v16bf ldsA_frag(const bf16* lds, int m, int g, int kbase) {
  v8bf lo = *(const v8bf*)(lds + m * LDS_STRIDE + kbase + g * 8);
  v8bf hi = *(const v8bf*)(lds + m * LDS_STRIDE + kbase + 16 + g * 8);
  v16bf a;
#pragma unroll
  for (int i = 0; i < 8; ++i) { a[i] = lo[i]; a[i + 8] = hi[i]; }
  return a;
}
// B (32x16 bf16, N=lane%16): element j -> K = (j<8?0:16) + (lane/16)*8 + (j%8) ; Wt is [fo][fi] so K contiguous
__device__ __forceinline__ v16bf gB_frag(const bf16* p) {
  v8bf lo = *(const v8bf*)(p);
  v8bf hi = *(const v8bf*)(p + 16);
  v16bf b;
#pragma unroll
  for (int i = 0; i < 8; ++i) { b[i] = lo[i]; b[i + 8] = hi[i]; }
  return b;
}

// ---------------- one fused layer on a 128-row tile ----------------
// mode 0: sin(w0*(x+b)) -> LDS bf16      (lout)
// mode 1: x+b           -> global bf16   (gchild = child base, row-guarded)
// mode 2: x+b           -> d_out f32 occ (gocc, row-guarded; occ_nin fixed 4096)
__device__ __forceinline__ void tile_layer(
    const bf16* __restrict__ lin, bf16* __restrict__ lout,
    const bf16* __restrict__ Wt, const float* __restrict__ bias,
    int fi, int fo, float w0, int mode,
    bf16* __restrict__ gchild, float* __restrict__ gocc,
    long row_base, long rows, int lane, int wave)
{
  const int g  = lane >> 4;   // lane half
  const int nl = lane & 15;   // local M / N index
  const int nct = fo >> 4;    // 16-col output tiles
  const int nkc = fi >> 5;    // K chunks of 32
  for (int ct = wave; ct < nct; ct += 8) {
    v8f acc[NRT];
#pragma unroll
    for (int rt = 0; rt < NRT; ++rt) acc[rt] = (v8f){};
    const int col0 = ct << 4;
    const bf16* wrow = Wt + (size_t)(col0 + nl) * fi + g * 8;
    for (int kc = 0; kc < nkc; ++kc) {
      v16bf b = gB_frag(wrow + kc * 32);          // one B frag, shared by 8 row tiles
#pragma unroll
      for (int rt = 0; rt < NRT; ++rt) {
        v16bf a = ldsA_frag(lin + rt * 16 * LDS_STRIDE, nl, g, kc * 32);
        acc[rt] = __builtin_amdgcn_wmma_f32_16x16x32_bf16(
            false, a, false, b, (short)0, acc[rt], false, false);
      }
    }
    const int col = col0 + nl;
    const float bv = bias[col];
    if (mode == 0) {
      const float kx = w0 * INV_2PI;      // fold w0 and 1/2pi into one FMA
      const float kb = bv * kx;
#pragma unroll
      for (int rt = 0; rt < NRT; ++rt)
#pragma unroll
        for (int v = 0; v < 8; ++v) {
          int m = rt * 16 + v + (g << 3);
          lout[m * LDS_STRIDE + col] =
              (bf16)__builtin_amdgcn_sinf(__builtin_fmaf(kx, acc[rt][v], kb));
        }
    } else if (mode == 1) {
#pragma unroll
      for (int rt = 0; rt < NRT; ++rt)
#pragma unroll
        for (int v = 0; v < 8; ++v) {
          long row = row_base + rt * 16 + v + (g << 3);
          if (row < rows) gchild[(size_t)row * fo + col] = (bf16)(acc[rt][v] + bv);
        }
    } else {
#pragma unroll
      for (int rt = 0; rt < NRT; ++rt)
#pragma unroll
        for (int v = 0; v < 8; ++v) {
          long row = row_base + rt * 16 + v + (g << 3);
          if (row < rows) {
            long bb = row >> 12, nn = row & 4095;   // occ_nin = 4096
            gocc[bb * OUT_STRIDE + LAT_OFF + nn * 16 + col] = acc[rt][v] + bv;
          }
        }
    }
  }
}

// ---------------- fused lodnet: 128 rows per block, whole MLP in LDS ----------------
__global__ __launch_bounds__(256)
void lodnet_kernel(const bf16* __restrict__ lat_in,
                   const bf16* Wb0t, const bf16* Wb1t, const bf16* Wb2t, const bf16* Wb3t,
                   const bf16* Wht,  const bf16* Wo0t, const bf16* Wo1t, const bf16* Wo2t,
                   const float* bb0, const float* bb1, const float* bb2, const float* bb3,
                   const float* bh,  const float* bo0, const float* bo1, const float* bo2,
                   bf16* __restrict__ childp, float* __restrict__ occp,
                   long rows, int compute_occ)
{
  __shared__ __align__(16) bf16 bufA[M_TILE * LDS_STRIDE];
  __shared__ __align__(16) bf16 bufB[M_TILE * LDS_STRIDE];
  const long row_base = (long)blockIdx.x * M_TILE;

  // stage 128x64 bf16 input latents (zero-fill past end)
  for (int t = threadIdx.x; t < M_TILE * 64; t += 256) {
    int m = t >> 6, c = t & 63;
    long row = row_base + m;
    bufA[m * LDS_STRIDE + c] = (row < rows) ? lat_in[(size_t)row * 64 + c] : (bf16)0.f;
  }
  __syncthreads();

  const int lane = threadIdx.x & 31, wave = threadIdx.x >> 5;
  tile_layer(bufA, bufB, Wb0t, bb0,  64, 512, 30.f, 0, nullptr, nullptr, row_base, rows, lane, wave);
  __syncthreads();
  tile_layer(bufB, bufA, Wb1t, bb1, 512, 512,  1.f, 0, nullptr, nullptr, row_base, rows, lane, wave);
  __syncthreads();
  tile_layer(bufA, bufB, Wb2t, bb2, 512, 512,  1.f, 0, nullptr, nullptr, row_base, rows, lane, wave);
  __syncthreads();
  tile_layer(bufB, bufA, Wb3t, bb3, 512, 512,  1.f, 0, nullptr, nullptr, row_base, rows, lane, wave);
  __syncthreads();
  // h in bufA. Child head straight to global (flat reshape to [8N][64] is contiguous).
  tile_layer(bufA, nullptr, Wht, bh, 512, 512, 1.f, 1,
             childp, nullptr, row_base, rows, lane, wave);
  if (compute_occ) {
    tile_layer(bufA, bufB, Wo0t, bo0, 512, 128, 1.f, 0, nullptr, nullptr, row_base, rows, lane, wave);
    __syncthreads();
    tile_layer(bufB, bufA, Wo1t, bo1, 128, 128, 1.f, 0, nullptr, nullptr, row_base, rows, lane, wave);
    __syncthreads();
    tile_layer(bufA, nullptr, Wo2t, bo2, 128, 16, 1.f, 2, nullptr, occp, row_base, rows, lane, wave);
  }
}

// ---------------- prep + gather ----------------
__global__ void transpose_bf16(const float* __restrict__ src, bf16* __restrict__ dst,
                               int fi, int fo) {
  int t = blockIdx.x * blockDim.x + threadIdx.x;
  if (t >= fi * fo) return;
  int o = t % fo, i = t / fo;                // src row-major [fi][fo]
  dst[(size_t)o * fi + i] = (bf16)src[t];    // dst [fo][fi]
}

__global__ void init_lat0(const float* __restrict__ feats, const int* __restrict__ idx,
                          bf16* __restrict__ lat) {
  int b = blockIdx.x, d = threadIdx.x;       // <<<32,64>>>
  lat[b * 64 + d] = (bf16)feats[(size_t)idx[b] * 64 + d];
}

// 16B per thread: 8 chunks of 8 bf16 per 64-elem row; sel/nchild are powers of two
__global__ void gather_kernel(const bf16* __restrict__ child, const int* __restrict__ ids,
                              bf16* __restrict__ lat_out, int sel_sh, int nch_sh) {
  size_t t = (size_t)blockIdx.x * blockDim.x + threadIdx.x;
  size_t total = (size_t)32 << (sel_sh + 3);
  if (t >= total) return;
  int c = (int)(t & 7);
  size_t r = t >> 3;
  size_t j = r & (((size_t)1 << sel_sh) - 1), b = r >> sel_sh;
  int id = ids[(b << sel_sh) + j];
  const uint4* src = (const uint4*)(child + (((size_t)b << nch_sh) + id) * 64);
  uint4* dst = (uint4*)(lat_out + r * 64);
  dst[c] = src[c];
}

__global__ void gather_out_kernel(const bf16* __restrict__ child, const int* __restrict__ ids,
                                  float* __restrict__ out) {
  size_t t = (size_t)blockIdx.x * blockDim.x + threadIdx.x;
  if (t >= (size_t)32 * 16384 * 8) return;
  int c = (int)(t & 7);
  size_t r = t >> 3, j = r & 16383, b = r >> 14;
  int id = ids[(b << 14) + j];
  v8bf s = *(const v8bf*)(child + ((b << 15) + (size_t)id) * 64 + c * 8);
  float* dst = out + b * (size_t)OUT_STRIDE + j * 64 + c * 8;
#pragma unroll
  for (int i = 0; i < 8; ++i) dst[i] = (float)s[i];
}

extern "C" void kernel_launch(void* const* d_in, const int* in_sizes, int n_in,
                              void* d_out, int out_size, void* d_ws, size_t ws_size,
                              hipStream_t stream) {
  const float* feats = (const float*)d_in[0];
  const float* Wb0 = (const float*)d_in[1];  const float* bb0 = (const float*)d_in[2];
  const float* Wb1 = (const float*)d_in[3];  const float* bb1 = (const float*)d_in[4];
  const float* Wb2 = (const float*)d_in[5];  const float* bb2 = (const float*)d_in[6];
  const float* Wb3 = (const float*)d_in[7];  const float* bb3 = (const float*)d_in[8];
  const float* Wh  = (const float*)d_in[9];  const float* bh  = (const float*)d_in[10];
  const float* Wo0 = (const float*)d_in[11]; const float* bo0 = (const float*)d_in[12];
  const float* Wo1 = (const float*)d_in[13]; const float* bo1 = (const float*)d_in[14];
  const float* Wo2 = (const float*)d_in[15]; const float* bo2 = (const float*)d_in[16];
  const int*   idx = (const int*)d_in[17];
  float* out = (float*)d_out;
  bf16*  ws  = (bf16*)d_ws;

  // workspace layout (bf16 element offsets)
  bf16* Wb0t = ws + 0;                  // 512x64
  bf16* Wb1t = ws + 32768;              // 512x512
  bf16* Wb2t = ws + 294912;
  bf16* Wb3t = ws + 557056;
  bf16* Wht  = ws + 819200;
  bf16* Wo0t = ws + 1081344;            // 128x512
  bf16* Wo1t = ws + 1146880;            // 128x128
  bf16* Wo2t = ws + 1163264;            // 16x128
  bf16* lat   = ws + 1179648;           // up to 32*16384*64 bf16
  bf16* child = ws + 1179648 + 33554432; // up to 32*32768*64 bf16

  struct { const float* s; bf16* d; int fi, fo; } W[8] = {
    {Wb0, Wb0t, 64, 512}, {Wb1, Wb1t, 512, 512}, {Wb2, Wb2t, 512, 512},
    {Wb3, Wb3t, 512, 512}, {Wh, Wht, 512, 512}, {Wo0, Wo0t, 512, 128},
    {Wo1, Wo1t, 128, 128}, {Wo2, Wo2t, 128, 16}};
  for (int i = 0; i < 8; ++i) {
    int n = W[i].fi * W[i].fo;
    transpose_bf16<<<(n + 255) / 256, 256, 0, stream>>>(W[i].s, W[i].d, W[i].fi, W[i].fo);
  }
  init_lat0<<<32, 64, 0, stream>>>(feats, idx, lat);

  const int SEL_SH[5] = {3, 6, 9, 12, 14};   // sel   = 8,64,512,4096,16384
  const int CH_SH[5]  = {3, 6, 9, 12, 15};   // child = 8,64,512,4096,32768
  long nin = 1;
  for (int lod = 0; lod < 5; ++lod) {
    long rows = 32 * nin;
    int grid = (int)((rows + M_TILE - 1) / M_TILE);
    lodnet_kernel<<<grid, 256, 0, stream>>>(
        lat, Wb0t, Wb1t, Wb2t, Wb3t, Wht, Wo0t, Wo1t, Wo2t,
        bb0, bb1, bb2, bb3, bh, bo0, bo1, bo2,
        child, out, rows, (lod == 4) ? 1 : 0);
    if (lod < 4) {
      size_t thr = (size_t)32 << (SEL_SH[lod] + 3);
      gather_kernel<<<(int)((thr + 255) / 256), 256, 0, stream>>>(
          child, (const int*)d_in[18 + lod], lat, SEL_SH[lod], CH_SH[lod]);
      nin = (long)1 << SEL_SH[lod];
    } else {
      size_t thr = (size_t)32 * 16384 * 8;
      gather_out_kernel<<<(int)((thr + 255) / 256), 256, 0, stream>>>(
          child, (const int*)d_in[22], out);
    }
  }
}
